// PerformerAttention_24352464569787
// MI455X (gfx1250) — matrix operand order repaired and checked by reading, not compiled, not run on hardware
//
#include <hip/hip_runtime.h>

// Problem constants (B,H,S,D,M) = (4,16,2048,64,256)
#define BB   4
#define HHH  16
#define SS   2048
#define DDD  64
#define MMM  256
#define EPSV 1e-3f
#define SCL  0.125f   /* D^{-1/2}; reference applies d^{-1/4} twice */

typedef __attribute__((ext_vector_type(16))) _Float16 v16h;
typedef __attribute__((ext_vector_type(8)))  _Float16 v8h;
typedef __attribute__((ext_vector_type(8)))  float    v8f;

// LDS row strides (halfs). All chosen so every fragment access is a 16B-aligned
// contiguous run per lane (=> ds_load_b128 / ds_store_b128).
#define PJ_LD  72    /* projL: [256][72]  natural [j][d]          */
#define KH_LD  72    /* kh/qh: [64][72]   row-major [s][d] (A op) */
#define VHT_LD 72    /* vhT:   [64][72]   transposed [e][s] (B op)*/
#define KPT_LD 72    /* kpT:   [256][72]  transposed [j][s]       */
#define QP_LD  264   /* qp:    [64][264]  row-major [s][j] (A op) */
#define CXT_LD 264   /* ctxT:  [64][264]  transposed [e][j] (B op)*/

#define WMMA_F16(a,b,c) __builtin_amdgcn_wmma_f32_16x16x32_f16(false,(a),false,(b),(short)0,(c),false,false)

// ---- fragment loaders (gfx1250 wave32 WMMA layouts, ISA 7.12.2) ----

// A (16x32 f16, MxK) from row-major [M][K]: lane<16 -> row m0+lane, K {0..7,16..23};
// lane>=16 -> K {8..15,24..31}. Two contiguous 8-half runs -> 2x ds_load_b128.
static __device__ inline v16h frag_a(const _Float16* p, int m0, int k0, int ld) {
  const int lane = threadIdx.x & 31;
  const _Float16* r = p + (m0 + (lane & 15)) * ld + (k0 + ((lane >> 4) << 3));
  v16h a;
#pragma unroll
  for (int i = 0; i < 8; ++i) { a[i] = r[i]; a[8 + i] = r[16 + i]; }
  return a;
}

// B (32x16 f16, KxN) from COLUMN-stored source ([N][K] row-major):
// col = n0 + lane&15; lanes<16 K {0..7,16..23}, lanes>=16 K {8..15,24..31}.
// Two contiguous 8-half runs per lane -> 2x ds_load_b128.
static __device__ inline v16h frag_bT(const _Float16* p, int k0, int n0, int ld) {
  const int lane = threadIdx.x & 31;
  const _Float16* r = p + (n0 + (lane & 15)) * ld + (k0 + ((lane >> 4) << 3));
  v16h b;
#pragma unroll
  for (int i = 0; i < 8; ++i) { b[i] = r[i]; b[8 + i] = r[16 + i]; }
  return b;
}

// =====================================================================
// Kernel 1: per (b,h) -> ctx[M][D] = kp^T @ (V*m),  ksum[M] = sum_s kp
// kp = relu((K * m * D^-1/2) @ proj^T) + eps, 64 S-rows per tile
// =====================================================================
__global__ __launch_bounds__(256) void perf_kv_ctx_kernel(
    const float* __restrict__ K, const float* __restrict__ V,
    const float* __restrict__ mask, const float* __restrict__ proj,
    float* __restrict__ ctx, float* __restrict__ ksum)
{
  extern __shared__ char smem[];
  _Float16* projL = (_Float16*)smem;            // [256][PJ_LD] natural [j][d]
  _Float16* kh    = projL + MMM * PJ_LD;        // [64][KH_LD]  [s][d]
  _Float16* vhT   = kh + 64 * KH_LD;            // [64][VHT_LD] [e][s]
  _Float16* kpT   = vhT + 64 * VHT_LD;          // [256][KPT_LD] [j][s]

  const int bh   = blockIdx.x;
  const int b    = bh >> 4;                     // H = 16
  const int tid  = threadIdx.x;
  const int w    = tid >> 5;
  const int lane = tid & 31;

  // proj -> LDS in natural [j][d] layout (coalesced float4 reads)
  {
    const float4* P4 = (const float4*)proj;
    for (int idx = tid; idx < MMM * 16; idx += 256) {
      int j = idx >> 4, c4 = idx & 15;
      float4 pv = P4[idx];
      int o = j * PJ_LD + c4 * 4;
      projL[o + 0] = (_Float16)pv.x; projL[o + 1] = (_Float16)pv.y;
      projL[o + 2] = (_Float16)pv.z; projL[o + 3] = (_Float16)pv.w;
    }
  }

  v8f acc[2][4];
#pragma unroll
  for (int mi = 0; mi < 2; ++mi)
#pragma unroll
    for (int e = 0; e < 4; ++e) acc[mi][e] = {};
  float ks = 0.f;

  const float4* K4 = (const float4*)(K + (size_t)bh * SS * DDD);
  const float4* V4 = (const float4*)(V + (size_t)bh * SS * DDD);

  for (int s0 = 0; s0 < SS; s0 += 64) {
    __syncthreads();
    if (s0 + 64 < SS) {  // uniform branch; warm caches for next tile
      __builtin_prefetch((const char*)&K4[(size_t)(s0 + 64) * 16 + (tid & 63) * 16], 0, 3);
      __builtin_prefetch((const char*)&V4[(size_t)(s0 + 64) * 16 + (tid & 63) * 16], 0, 3);
    }
    // ---- stage: kh[s][d] = K*m*scl ; vhT[e][s] = V*m (transposed for B frags) ----
    for (int idx = tid; idx < 64 * 16; idx += 256) {
      int r = idx >> 4, c4 = idx & 15;
      float  m  = mask[(size_t)b * SS + s0 + r];
      float4 kv = K4[(size_t)(s0 + r) * 16 + c4];
      float4 vv = V4[(size_t)(s0 + r) * 16 + c4];
      float  km = m * SCL;
      int o = r * KH_LD + c4 * 4;
      kh[o + 0] = (_Float16)(kv.x * km); kh[o + 1] = (_Float16)(kv.y * km);
      kh[o + 2] = (_Float16)(kv.z * km); kh[o + 3] = (_Float16)(kv.w * km);
      int e = c4 * 4;
      vhT[(e + 0) * VHT_LD + r] = (_Float16)(vv.x * m);
      vhT[(e + 1) * VHT_LD + r] = (_Float16)(vv.y * m);
      vhT[(e + 2) * VHT_LD + r] = (_Float16)(vv.z * m);
      vhT[(e + 3) * VHT_LD + r] = (_Float16)(vv.w * m);
    }
    __syncthreads();

    // ---- GEMM1: kpT[j][s] = relu(kh @ projL^T) + eps ; A frags hoisted ----
    {
      const int sblk  = (w & 3) * 16;
      const int mbase = (w >> 2) * 128;
      const v16h a0 = frag_a(kh, sblk, 0, KH_LD);
      const v16h a1 = frag_a(kh, sblk, 32, KH_LD);
      const int n = lane & 15, rb = (lane >> 4) << 3;
#pragma unroll
      for (int mb = 0; mb < 8; ++mb) {
        int m0 = mbase + mb * 16;
        v8f c = {};
        c = WMMA_F16(a0, frag_bT(projL, 0, m0, PJ_LD), c);
        c = WMMA_F16(a1, frag_bT(projL, 32, m0, PJ_LD), c);
        v8h h;
#pragma unroll
        for (int i = 0; i < 8; ++i) h[i] = (_Float16)(fmaxf(c[i], 0.f) + EPSV);
        // lane's 8 values are consecutive S rows of one column -> contiguous store
        *(v8h*)(kpT + (m0 + n) * KPT_LD + sblk + rb) = h;
      }
    }
    __syncthreads();

    // ---- ksum partial: thread t owns row j=t of kpT (contiguous b128 reads) ----
    {
      float s = 0.f;
#pragma unroll
      for (int r8 = 0; r8 < 8; ++r8) {
        v8h x = *(const v8h*)(kpT + tid * KPT_LD + r8 * 8);
#pragma unroll
        for (int i = 0; i < 8; ++i) s += (float)x[i];
      }
      ks += s;
    }

    // ---- GEMM2: ctx += kp^T @ vh ; A natural from kpT, B from vhT; all hoisted ----
#pragma unroll
    for (int kk = 0; kk < 64; kk += 32) {
      const v16h a0 = frag_a(kpT, w * 32,      kk, KPT_LD);
      const v16h a1 = frag_a(kpT, w * 32 + 16, kk, KPT_LD);
      const v16h b0 = frag_bT(vhT, kk,  0, VHT_LD);
      const v16h b1 = frag_bT(vhT, kk, 16, VHT_LD);
      const v16h b2 = frag_bT(vhT, kk, 32, VHT_LD);
      const v16h b3 = frag_bT(vhT, kk, 48, VHT_LD);
      acc[0][0] = WMMA_F16(a0, b0, acc[0][0]);
      acc[0][1] = WMMA_F16(a0, b1, acc[0][1]);
      acc[0][2] = WMMA_F16(a0, b2, acc[0][2]);
      acc[0][3] = WMMA_F16(a0, b3, acc[0][3]);
      acc[1][0] = WMMA_F16(a1, b0, acc[1][0]);
      acc[1][1] = WMMA_F16(a1, b1, acc[1][1]);
      acc[1][2] = WMMA_F16(a1, b2, acc[1][2]);
      acc[1][3] = WMMA_F16(a1, b3, acc[1][3]);
    }
  }

  // ---- store ksum and ctx ----
  ksum[(size_t)bh * MMM + tid] = ks;
  float* cdst = ctx + (size_t)bh * MMM * DDD;
  const int n = lane & 15, rb = (lane >> 4) << 3;
#pragma unroll
  for (int mi = 0; mi < 2; ++mi) {
    int m0 = w * 32 + mi * 16;
#pragma unroll
    for (int e = 0; e < 4; ++e)
#pragma unroll
      for (int i = 0; i < 8; ++i)
        cdst[(size_t)(m0 + rb + i) * DDD + e * 16 + n] = acc[mi][e][i];
  }
}

// =====================================================================
// Kernel 2: per (b,h, 64-row S tile):
//   qp = relu((Q*D^-1/2) @ proj^T)+eps ; den = qp @ ksum ; out = (qp@ctx)/den
// ctx (4 MiB) + proj are L2-resident -> reloads don't hit HBM.
// =====================================================================
__global__ __launch_bounds__(256) void perf_q_out_kernel(
    const float* __restrict__ Q, const float* __restrict__ proj,
    const float* __restrict__ ctx, const float* __restrict__ ksum,
    float* __restrict__ out)
{
  extern __shared__ char smem[];
  float*    ksumL = (float*)smem;                       // [256] f32
  float*    den   = ksumL + MMM;                        // [64]  f32
  _Float16* projL = (_Float16*)(den + 64);              // [256][PJ_LD]
  _Float16* qh    = projL + MMM * PJ_LD;                // [64][KH_LD]
  _Float16* qp    = qh + 64 * KH_LD;                    // [64][QP_LD] [s][j]
  _Float16* ctxT  = qp + 64 * QP_LD;                    // [64][CXT_LD] [e][j]

  const int bh   = blockIdx.y;
  const int s0   = blockIdx.x * 64;
  const int tid  = threadIdx.x;
  const int w    = tid >> 5;
  const int lane = tid & 31;

  {
    const float4* P4 = (const float4*)proj;
    for (int idx = tid; idx < MMM * 16; idx += 256) {
      int j = idx >> 4, c4 = idx & 15;
      float4 pv = P4[idx];
      int o = j * PJ_LD + c4 * 4;
      projL[o + 0] = (_Float16)pv.x; projL[o + 1] = (_Float16)pv.y;
      projL[o + 2] = (_Float16)pv.z; projL[o + 3] = (_Float16)pv.w;
    }
  }
  ksumL[tid] = ksum[(size_t)bh * MMM + tid];
  if (tid < 64) den[tid] = 0.f;

  const float4* Q4 = (const float4*)(Q + (size_t)bh * SS * DDD);
  for (int idx = tid; idx < 64 * 16; idx += 256) {
    int r = idx >> 4, c4 = idx & 15;
    float4 qv = Q4[(size_t)(s0 + r) * 16 + c4];
    int o = r * KH_LD + c4 * 4;
    qh[o + 0] = (_Float16)(qv.x * SCL); qh[o + 1] = (_Float16)(qv.y * SCL);
    qh[o + 2] = (_Float16)(qv.z * SCL); qh[o + 3] = (_Float16)(qv.w * SCL);
  }
  // ctx[j][e] -> ctxT[e][j] (transposed for contiguous B frags)
  {
    const float4* C4 = (const float4*)(ctx + (size_t)bh * MMM * DDD);
    for (int idx = tid; idx < MMM * 16; idx += 256) {
      int j = idx >> 4, c4 = idx & 15;
      float4 cv = C4[idx];
      int e = c4 * 4;
      ctxT[(e + 0) * CXT_LD + j] = (_Float16)cv.x;
      ctxT[(e + 1) * CXT_LD + j] = (_Float16)cv.y;
      ctxT[(e + 2) * CXT_LD + j] = (_Float16)cv.z;
      ctxT[(e + 3) * CXT_LD + j] = (_Float16)cv.w;
    }
  }
  __syncthreads();

  // ---- GEMM1: qp[s][j] = relu(qh @ projL^T) + eps ----
  {
    const int sblk  = (w & 3) * 16;
    const int mbase = (w >> 2) * 128;
    const v16h a0 = frag_a(qh, sblk, 0, KH_LD);
    const v16h a1 = frag_a(qh, sblk, 32, KH_LD);
    const int n = lane & 15, rb = (lane >> 4) << 3;
#pragma unroll
    for (int mb = 0; mb < 8; ++mb) {
      int m0 = mbase + mb * 16;
      v8f c = {};
      c = WMMA_F16(a0, frag_bT(projL, 0, m0, PJ_LD), c);
      c = WMMA_F16(a1, frag_bT(projL, 32, m0, PJ_LD), c);
#pragma unroll
      for (int i = 0; i < 8; ++i) {
        float x = fmaxf(c[i], 0.f) + EPSV;
        qp[(sblk + rb + i) * QP_LD + m0 + n] = (_Float16)x;  // A op stays [s][j]
      }
    }
  }
  __syncthreads();

  // ---- den[row] = qp[row,:] . ksum (4 threads/row, b128 reads, LDS f32 atomics) ----
  {
    int row = tid >> 2, part = (tid & 3) * 64;
    float s = 0.f;
#pragma unroll
    for (int j8 = 0; j8 < 8; ++j8) {
      v8h x = *(const v8h*)(qp + row * QP_LD + part + j8 * 8);
#pragma unroll
      for (int i = 0; i < 8; ++i) s += (float)x[i] * ksumL[part + j8 * 8 + i];
    }
    atomicAdd(&den[row], s);
  }
  __syncthreads();

  // ---- GEMM2: out = (qp @ ctx) * (1/den) ; 2 tiles/wave sharing the A frag ----
  {
    float* odst = out + ((size_t)bh * SS + s0) * DDD;
    const int sblk = (w >> 1) * 16;
    const int e0   = (w & 1) * 32;
    v8f c0 = {}, c1 = {};
#pragma unroll
    for (int kk = 0; kk < MMM; kk += 32) {
      const v16h a = frag_a(qp, sblk, kk, QP_LD);
      c0 = WMMA_F16(a, frag_bT(ctxT, kk, e0,      CXT_LD), c0);
      c1 = WMMA_F16(a, frag_bT(ctxT, kk, e0 + 16, CXT_LD), c1);
    }
    const int n = lane & 15, rb = (lane >> 4) << 3;
#pragma unroll
    for (int i = 0; i < 8; ++i) {
      int row = sblk + rb + i;
      float dinv = 1.0f / den[row];
      odst[(size_t)row * DDD + e0 + n]      = c0[i] * dinv;
      odst[(size_t)row * DDD + e0 + 16 + n] = c1[i] * dinv;
    }
  }
}

extern "C" void kernel_launch(void* const* d_in, const int* in_sizes, int n_in,
                              void* d_out, int out_size, void* d_ws, size_t ws_size,
                              hipStream_t stream) {
  (void)in_sizes; (void)n_in; (void)out_size; (void)ws_size;
  const float* Q    = (const float*)d_in[0];
  const float* K    = (const float*)d_in[1];
  const float* V    = (const float*)d_in[2];
  const float* mask = (const float*)d_in[3];
  const float* proj = (const float*)d_in[4];
  float* out  = (float*)d_out;
  float* ctx  = (float*)d_ws;                               // 64*256*64 f32 = 4 MiB
  float* ksum = ctx + (size_t)BB * HHH * MMM * DDD;         // 64*256 f32

  const size_t lds1 = (size_t)(MMM * PJ_LD + 64 * KH_LD + 64 * VHT_LD + MMM * KPT_LD) * 2;
  const size_t lds2 = (size_t)(MMM + 64) * 4 +
                      (size_t)(MMM * PJ_LD + 64 * KH_LD + 64 * QP_LD + 64 * CXT_LD) * 2;

  perf_kv_ctx_kernel<<<BB * HHH, 256, lds1, stream>>>(K, V, mask, proj, ctx, ksum);
  perf_q_out_kernel<<<dim3(SS / 64, BB * HHH), 256, lds2, stream>>>(Q, proj, ctx, ksum, out);
}
